// Model_55216099557796
// MI455X (gfx1250) — matrix operand order repaired and checked by reading, compiled, not run on hardware
//
#include <hip/hip_runtime.h>

// ---------------------------------------------------------------------------
// GCN forward on MI455X (gfx1250), bf16 WMMA path.
//   N=16384, F_IN=256, F_HID=128, F_OUT=16
//   S1  = X @ W1                      (stored transposed, bf16)
//   H1  = relu(adj @ S1 + b1)         (bf16)
//   S2  = H1 @ W2                     (stored transposed, bf16)
//   out = softmax(adj @ S2 + b2)      (f32)
// ---------------------------------------------------------------------------

typedef __attribute__((ext_vector_type(16))) __bf16 v16bf;
typedef __attribute__((ext_vector_type(8)))  __bf16 v8bf;
typedef __attribute__((ext_vector_type(8)))  float  v8f;

#define NROW 16384
#define FIN  256
#define FHID 128
#define FOUT 16
#define KSTEP 32

__device__ __forceinline__ v8f zero8() {
    v8f z;
#pragma unroll
    for (int i = 0; i < 8; ++i) z[i] = 0.0f;
    return z;
}

// A-fragment (16x32 bf16, M x K): lane holds row M=lane%16; group g=lane/16.
// element e: K = 16*g + e  -> one contiguous 32B load.
__device__ __forceinline__ v16bf load_a_frag(const __bf16* __restrict__ row, int g) {
    return *(const v16bf*)(row + 16 * g);
}

// B-fragment (32x16 bf16, K x N) from an N-major staging buffer:
// lane holds column N=lane%16, K = 16*g + e -> one contiguous 32B read.
__device__ __forceinline__ v16bf load_b_frag(const __bf16* __restrict__ colKmajor, int g) {
    return *(const v16bf*)(colKmajor + 16 * g);
}

// ---------------------------------------------------------------------------
// Kernel 1: S1^T[128][16384] = (X[16384][256] @ W1[256][128])^T   (bf16 out)
// block = 256 thr (8 waves), each block does 128 rows, each wave 16 rows x 128
// ---------------------------------------------------------------------------
__global__ __launch_bounds__(256) void gcn_xw1(const float* __restrict__ x,
                                               const float* __restrict__ W1,
                                               __bf16* __restrict__ s1t) {
    __shared__ __bf16 A_lds[128][KSTEP];
    __shared__ __bf16 B_lds[FHID][KSTEP];

    const int t    = threadIdx.x;
    const int lane = t & 31;
    const int w    = t >> 5;
    const int g    = lane >> 4;
    const int ln   = lane & 15;
    const int m_base = blockIdx.x * 128;

    v8f acc[8];
#pragma unroll
    for (int i = 0; i < 8; ++i) acc[i] = zero8();

    for (int k0 = 0; k0 < FIN; k0 += KSTEP) {
        __syncthreads();
        {   // stage A tile (128 x 32 f32 -> bf16)
            const int r0 = t >> 3;
            const int c4 = (t & 7) * 4;
#pragma unroll
            for (int rr = 0; rr < 128; rr += 32) {
                const float4 v = *(const float4*)(x + (size_t)(m_base + r0 + rr) * FIN + k0 + c4);
                A_lds[r0 + rr][c4 + 0] = (__bf16)v.x;
                A_lds[r0 + rr][c4 + 1] = (__bf16)v.y;
                A_lds[r0 + rr][c4 + 2] = (__bf16)v.z;
                A_lds[r0 + rr][c4 + 3] = (__bf16)v.w;
            }
        }
        {   // stage B tile N-major: B_lds[n][k] = W1[k0+k][n]
            const int n    = t >> 1;
            const int koff = (t & 1) * 16;
#pragma unroll
            for (int i = 0; i < 16; ++i)
                B_lds[n][koff + i] = (__bf16)W1[(size_t)(k0 + koff + i) * FHID + n];
        }
        __syncthreads();

        const v16bf afrag = load_a_frag(&A_lds[w * 16 + ln][0], g);
#pragma unroll
        for (int nt = 0; nt < 8; ++nt) {
            const v16bf bfrag = load_b_frag(&B_lds[nt * 16 + ln][0], g);
            acc[nt] = __builtin_amdgcn_wmma_f32_16x16x32_bf16(
                false, afrag, false, bfrag, (short)0, acc[nt], false, false);
        }
    }

    // store transposed: s1t[n][m]
#pragma unroll
    for (int nt = 0; nt < 8; ++nt) {
        const int n = nt * 16 + ln;
#pragma unroll
        for (int r = 0; r < 8; ++r) {
            const int m = m_base + w * 16 + r + 8 * g;
            s1t[(size_t)n * NROW + m] = (__bf16)acc[nt][r];
        }
    }
}

// ---------------------------------------------------------------------------
// Kernel 2: H1[16384][128] = relu(adj @ S1 + b1)        (bf16 out)
// The dominant GEMM: M=N_nodes, K=16384, N=128.
// ---------------------------------------------------------------------------
__global__ __launch_bounds__(256) void gcn_adj_h1(const float* __restrict__ adj,
                                                  const __bf16* __restrict__ s1t,
                                                  const float* __restrict__ b1,
                                                  __bf16* __restrict__ h1) {
    __shared__ __bf16 A_lds[128][KSTEP];   // adj tile, bf16
    __shared__ __bf16 B_lds[FHID][KSTEP];  // S1^T tile, N-major

    const int t    = threadIdx.x;
    const int lane = t & 31;
    const int w    = t >> 5;
    const int g    = lane >> 4;
    const int ln   = lane & 15;
    const int m_base = blockIdx.x * 128;

    v8f acc[8];
#pragma unroll
    for (int i = 0; i < 8; ++i) acc[i] = zero8();

    const int r0 = t >> 3;
    const int c4 = (t & 7) * 4;
    const int nB    = t >> 1;
    const int koffB = (t & 1) * 16;

    for (int k0 = 0; k0 < NROW; k0 += KSTEP) {
        __syncthreads();
        // stage A tile (128x32 f32 -> bf16) with prefetch of the next tile
#pragma unroll
        for (int rr = 0; rr < 128; rr += 32) {
            const float* src = adj + (size_t)(m_base + r0 + rr) * NROW + k0 + c4;
            const float4 v = *(const float4*)src;
            __builtin_prefetch(src + KSTEP, 0, 0);
            A_lds[r0 + rr][c4 + 0] = (__bf16)v.x;
            A_lds[r0 + rr][c4 + 1] = (__bf16)v.y;
            A_lds[r0 + rr][c4 + 2] = (__bf16)v.z;
            A_lds[r0 + rr][c4 + 3] = (__bf16)v.w;
        }
        // stage B tile: contiguous 32B copy per thread (S1^T is K-contiguous)
        *(v16bf*)&B_lds[nB][koffB] = *(const v16bf*)(s1t + (size_t)nB * NROW + k0 + koffB);
        __syncthreads();

        const v16bf afrag = load_a_frag(&A_lds[w * 16 + ln][0], g);
#pragma unroll
        for (int nt = 0; nt < 8; ++nt) {
            const v16bf bfrag = load_b_frag(&B_lds[nt * 16 + ln][0], g);
            acc[nt] = __builtin_amdgcn_wmma_f32_16x16x32_bf16(
                false, afrag, false, bfrag, (short)0, acc[nt], false, false);
        }
    }

    // epilogue: + b1, relu, bf16 store (row-major, K-contiguous for kernel 3)
#pragma unroll
    for (int nt = 0; nt < 8; ++nt) {
        const int n = nt * 16 + ln;
        const float bias = b1[n];
#pragma unroll
        for (int r = 0; r < 8; ++r) {
            const int m = m_base + w * 16 + r + 8 * g;
            float v = acc[nt][r] + bias;
            v = v > 0.0f ? v : 0.0f;
            h1[(size_t)m * FHID + n] = (__bf16)v;
        }
    }
}

// ---------------------------------------------------------------------------
// Kernel 3: S2^T[16][16384] = (H1 @ W2)^T   (bf16 out);  K=128, N=16
// ---------------------------------------------------------------------------
__global__ __launch_bounds__(256) void gcn_h1w2(const __bf16* __restrict__ h1,
                                                const float* __restrict__ W2,
                                                __bf16* __restrict__ s2t) {
    __shared__ __bf16 W2T[FOUT][FHID];  // N-major, K-contiguous

    const int t    = threadIdx.x;
    const int lane = t & 31;
    const int w    = t >> 5;
    const int g    = lane >> 4;
    const int ln   = lane & 15;
    const int m_base = blockIdx.x * 128;

    {   // stage W2^T once: W2T[n][k] = W2[k][n]
        const int n  = t & 15;
        const int kb = (t >> 4) * 8;
#pragma unroll
        for (int i = 0; i < 8; ++i)
            W2T[n][kb + i] = (__bf16)W2[(size_t)(kb + i) * FOUT + n];
    }
    __syncthreads();

    v8f acc = zero8();
    const int m = m_base + w * 16 + ln;  // this lane's A row
#pragma unroll
    for (int k0 = 0; k0 < FHID; k0 += KSTEP) {
        const v16bf afrag = load_a_frag(h1 + (size_t)m * FHID + k0, g);
        const v16bf bfrag = load_b_frag(&W2T[ln][k0], g);
        acc = __builtin_amdgcn_wmma_f32_16x16x32_bf16(
            false, afrag, false, bfrag, (short)0, acc, false, false);
    }

#pragma unroll
    for (int r = 0; r < 8; ++r) {
        const int mm = m_base + w * 16 + r + 8 * g;
        s2t[(size_t)ln * NROW + mm] = (__bf16)acc[r];
    }
}

// ---------------------------------------------------------------------------
// Kernel 4: out = softmax(adj @ S2 + b2, axis=1)   (f32 out); N=16
// block does 256 rows; each wave 32 rows (2 M-tiles). Softmax fused via
// 16-lane shuffle reductions over the C-layout N dimension.
// ---------------------------------------------------------------------------
__global__ __launch_bounds__(256) void gcn_adj_out(const float* __restrict__ adj,
                                                   const __bf16* __restrict__ s2t,
                                                   const float* __restrict__ b2,
                                                   float* __restrict__ out) {
    __shared__ __bf16 A_lds[256][KSTEP];
    __shared__ __bf16 B_lds[FOUT][KSTEP];

    const int t    = threadIdx.x;
    const int lane = t & 31;
    const int w    = t >> 5;
    const int g    = lane >> 4;
    const int ln   = lane & 15;
    const int m_base = blockIdx.x * 256;

    v8f acc[2] = {zero8(), zero8()};

    const int r0 = t >> 3;
    const int c4 = (t & 7) * 4;

    for (int k0 = 0; k0 < NROW; k0 += KSTEP) {
        __syncthreads();
#pragma unroll
        for (int rr = 0; rr < 256; rr += 32) {
            const float* src = adj + (size_t)(m_base + r0 + rr) * NROW + k0 + c4;
            const float4 v = *(const float4*)src;
            __builtin_prefetch(src + KSTEP, 0, 0);
            A_lds[r0 + rr][c4 + 0] = (__bf16)v.x;
            A_lds[r0 + rr][c4 + 1] = (__bf16)v.y;
            A_lds[r0 + rr][c4 + 2] = (__bf16)v.z;
            A_lds[r0 + rr][c4 + 3] = (__bf16)v.w;
        }
        if (t < 32) {  // 16 x 32 bf16 B tile
            const int n    = t >> 1;
            const int koff = (t & 1) * 16;
            *(v16bf*)&B_lds[n][koff] = *(const v16bf*)(s2t + (size_t)n * NROW + k0 + koff);
        }
        __syncthreads();

        const v16bf bfrag = load_b_frag(&B_lds[ln][0], g);
#pragma unroll
        for (int mt = 0; mt < 2; ++mt) {
            const v16bf afrag = load_a_frag(&A_lds[w * 32 + mt * 16 + ln][0], g);
            acc[mt] = __builtin_amdgcn_wmma_f32_16x16x32_bf16(
                false, afrag, false, bfrag, (short)0, acc[mt], false, false);
        }
    }

    // fused bias + row softmax (16 cols live in 16 consecutive lanes)
    const float bias = b2[ln];
#pragma unroll
    for (int mt = 0; mt < 2; ++mt) {
#pragma unroll
        for (int r = 0; r < 8; ++r) {
            const float v = acc[mt][r] + bias;
            float mx = v;
#pragma unroll
            for (int off = 8; off >= 1; off >>= 1)
                mx = fmaxf(mx, __shfl_xor(mx, off, 32));
            const float e = __expf(v - mx);
            float s = e;
#pragma unroll
            for (int off = 8; off >= 1; off >>= 1)
                s += __shfl_xor(s, off, 32);
            const int m = m_base + w * 32 + mt * 16 + r + 8 * g;
            out[(size_t)m * FOUT + ln] = e / s;
        }
    }
}

// ---------------------------------------------------------------------------
extern "C" void kernel_launch(void* const* d_in, const int* in_sizes, int n_in,
                              void* d_out, int out_size, void* d_ws, size_t ws_size,
                              hipStream_t stream) {
    (void)in_sizes; (void)n_in; (void)out_size; (void)ws_size;
    const float* in_feat = (const float*)d_in[0];
    const float* adj     = (const float*)d_in[1];
    const float* W1      = (const float*)d_in[2];
    const float* b1      = (const float*)d_in[3];
    const float* W2      = (const float*)d_in[4];
    const float* b2      = (const float*)d_in[5];
    float* out = (float*)d_out;

    char* ws = (char*)d_ws;
    __bf16* s1t = (__bf16*)(ws);                                       // 128 x 16384 bf16 = 4 MiB
    __bf16* h1  = (__bf16*)(ws + (size_t)FHID * NROW * 2);             // 16384 x 128 bf16 = 4 MiB
    __bf16* s2t = (__bf16*)(ws + (size_t)FHID * NROW * 4);             // 16 x 16384 bf16  = 512 KiB

    gcn_xw1   <<<NROW / 128, 256, 0, stream>>>(in_feat, W1, s1t);
    gcn_adj_h1<<<NROW / 128, 256, 0, stream>>>(adj, s1t, b1, h1);
    gcn_h1w2  <<<NROW / 128, 256, 0, stream>>>(h1, W2, s2t);
    gcn_adj_out<<<NROW / 256, 256, 0, stream>>>(adj, s2t, b2, out);
}
